// LorentzAttention_77163382439945
// MI455X (gfx1250) — matrix operand order repaired and checked by reading, not compiled, not run on hardware
//
#include <hip/hip_runtime.h>
#include <hip/hip_bf16.h>

typedef _Float16 half8  __attribute__((ext_vector_type(8)));
typedef _Float16 v16h   __attribute__((ext_vector_type(16)));
typedef float    v8f    __attribute__((ext_vector_type(8)));
typedef unsigned int ui32x4 __attribute__((ext_vector_type(4)));
typedef int          i32x8  __attribute__((ext_vector_type(8)));
typedef int          i32x4  __attribute__((ext_vector_type(4)));

#define EPSF 1e-6f

static __device__ __forceinline__ v16h cat8(half8 a, half8 b) {
  return __builtin_shufflevector(a, b, 0,1,2,3,4,5,6,7,8,9,10,11,12,13,14,15);
}

static __device__ __forceinline__ v8f wmma_f16(v16h a, v16h b, v8f c) {
  // D = A(16x32 f16) x B(32x16 f16) + C(16x16 f32)
  return __builtin_amdgcn_wmma_f32_16x16x32_f16(
      /*neg_a=*/false, a, /*neg_b=*/false, b,
      /*c_mod=*/(short)0, c, /*reuse_a=*/false, /*reuse_b=*/false);
}

// TDM: 2-D tile load global->LDS via D# descriptor (CDNA5 ISA).
// tileW elems per row, tileH rows, rowStride elems, 2-byte elements.
// 6-arg builtin form (clang-23 / therock-10.0 headers).
static __device__ __forceinline__ void tdm_load_2d_f16(unsigned lds_addr,
                                                       const void* gptr,
                                                       unsigned tileW, unsigned tileH,
                                                       unsigned rowStride,
                                                       unsigned tdim0, unsigned tdim1) {
  unsigned long long ga = (unsigned long long)(uintptr_t)gptr;
  ui32x4 g0;
  g0[0] = 1u;                                            // count=1, user mode
  g0[1] = lds_addr;                                      // lds_addr [63:32]
  g0[2] = (unsigned)(ga & 0xffffffffu);                  // global_addr lo
  g0[3] = (unsigned)((ga >> 32) & 0x1ffffffu) | (2u << 30);  // addr hi | type=2
  i32x8 g1;
  g1[0] = (int)(1u << 16);                               // data_size=1 (2B)
  g1[1] = (int)((tdim0 & 0xffffu) << 16);                // tensor_dim0 lo16
  g1[2] = (int)((tdim0 >> 16) | ((tdim1 & 0xffffu) << 16));
  g1[3] = (int)((tdim1 >> 16) | (tileW << 16));          // tile_dim0
  g1[4] = (int)(tileH & 0xffffu);                        // tile_dim1 (tile_dim2=0)
  g1[5] = (int)rowStride;                                // tensor_dim0_stride lo32
  g1[6] = 0;                                             // stride hi / dim1_stride
  g1[7] = 0;
  i32x4 z4 = {0, 0, 0, 0};
  i32x8 z8 = {0, 0, 0, 0, 0, 0, 0, 0};
  __builtin_amdgcn_tensor_load_to_lds(g0, g1, z4, z4, z8, 0);
}

// ---------------------------------------------------------------------------
// Kernel 1: v = log_map(mu, x), mu = e0.  One block per token (2048 blocks).
// ---------------------------------------------------------------------------
__global__ __launch_bounds__(256) void k_logmap(const float* __restrict__ x,
                                                _Float16* __restrict__ vh) {
  const int d = 1024;
  const int t = blockIdx.x;
  const float* xr = x + (size_t)t * d;
  __shared__ float red[256];

  float s = 0.f;
  for (int j = threadIdx.x; j < d; j += 256) {
    if (j >= 1) {
      float sp = fminf(fmaxf(xr[j], -8.f), 8.f);
      s += sp * sp;
    }
  }
  red[threadIdx.x] = s;
  __syncthreads();
  for (int off = 128; off > 0; off >>= 1) {
    if (threadIdx.x < off) red[threadIdx.x] += red[threadIdx.x + off];
    __syncthreads();
  }
  const float S  = red[0];
  const float y0 = sqrtf(1.f + S + EPSF);
  const float xy = fminf(-y0, -(1.f + EPSF));
  const float z  = fmaxf(-xy, 1.f + EPSF);
  const float dg = fmaxf(logf(z + sqrtf(z * z - 1.f)), 0.001f);  // arccosh
  const float d0 = y0 + xy;                                      // direc[0]
  const float dn = sqrtf(fmaxf(S - d0 * d0, EPSF));
  const float sc = dg / (dn + EPSF);

  _Float16* vr = vh + (size_t)t * d;
  for (int j = threadIdx.x; j < d; j += 256) {
    float val = (j == 0) ? sc * d0 : sc * fminf(fmaxf(xr[j], -8.f), 8.f);
    vr[j] = (_Float16)val;
  }
}

// ---------------------------------------------------------------------------
// Kernel 2: C[m][n] = sum_k A[m][k] * W[n][k]   (A @ W^T)
// A: f16 [M][K] staged via TDM (TENSORcnt); W: f32 [N][K] converted in LDS.
// Tile 64x64, BK=32, 128 threads = 4 waves.
// ---------------------------------------------------------------------------
#define GBM 64
#define GBN 64
#define GBK 32
__global__ __launch_bounds__(128) void k_gemm(const _Float16* __restrict__ A,
                                              const float* __restrict__ W,
                                              _Float16* __restrict__ C,
                                              int M, int N, int K) {
  const int nTilesN = N / GBN;
  const int m0 = (blockIdx.x / nTilesN) * GBM;
  const int n0 = (blockIdx.x % nTilesN) * GBN;

  __shared__ __align__(16) _Float16 As[GBM][GBK];  // 4 KB, filled by TDM
  __shared__ __align__(16) _Float16 Ws[GBN][GBK];  // 4 KB, rows = n

  const int tid  = threadIdx.x;
  const int lane = tid & 31;
  const int wave = tid >> 5;        // 0..3 -> M strip
  const int mrow = lane & 15;
  const int hi   = lane >> 4;       // 0/1

  const unsigned asLds = (unsigned)(uintptr_t)&As[0][0];

  v8f acc[4] = {};

  for (int k0 = 0; k0 < K; k0 += GBK) {
    __syncthreads();
    // ---- A tile: Tensor Data Mover, one descriptor issued by wave 0
    if (wave == 0) {
      tdm_load_2d_f16(asLds, &A[(size_t)m0 * K + k0],
                      GBK, GBM, (unsigned)K, (unsigned)K, (unsigned)M);
      __builtin_amdgcn_s_wait_tensorcnt(0);
    }
    // ---- W tile: 64x32, f32 -> f16 convert, 16 floats per thread
    {
      int row = tid >> 1;
      int kk  = (tid & 1) * 16;
      const float4* wsrc = (const float4*)&W[(size_t)(n0 + row) * K + k0 + kk];
      #pragma unroll
      for (int j = 0; j < 4; ++j) {
        float4 f = wsrc[j];
        Ws[row][kk + j * 4 + 0] = (_Float16)f.x;
        Ws[row][kk + j * 4 + 1] = (_Float16)f.y;
        Ws[row][kk + j * 4 + 2] = (_Float16)f.z;
        Ws[row][kk + j * 4 + 3] = (_Float16)f.w;
      }
      if (k0 + GBK < K) {  // prefetch next W slab -> global_prefetch_b8
        __builtin_prefetch(&W[(size_t)(n0 + row) * K + k0 + GBK], 0, 0);
      }
    }
    __syncthreads();

    // A fragment (16x32, layout per ISA 7.12.2)
    const int am   = wave * 16 + mrow;
    const int koff = hi * 8;
    v16h af = cat8(*(const half8*)&As[am][koff],
                   *(const half8*)&As[am][16 + koff]);
    // batch all 4 B fragments, then issue WMMAs back-to-back
    v16h bf[4];
    #pragma unroll
    for (int nb = 0; nb < 4; ++nb) {
      const int bn = nb * 16 + mrow;
      const int kb = hi * 8;
      bf[nb] = cat8(*(const half8*)&Ws[bn][kb],
                    *(const half8*)&Ws[bn][kb + 16]);
    }
    #pragma unroll
    for (int nb = 0; nb < 4; ++nb) acc[nb] = wmma_f16(af, bf[nb], acc[nb]);
  }

  // epilogue: C layout (lanes 0-15: M=r, lanes 16-31: M=8+r)
  #pragma unroll
  for (int nb = 0; nb < 4; ++nb)
    #pragma unroll
    for (int r = 0; r < 8; ++r) {
      int row = m0 + wave * 16 + hi * 8 + r;
      int col = n0 + nb * 16 + mrow;
      C[(size_t)row * N + col] = (_Float16)acc[nb][r];
    }
}

// ---------------------------------------------------------------------------
// Kernel 3: per-head projection + layout transform.
// mode 0: clip/recompute x0 (K); mode 1: same + negate comp0 (Q); mode 2: copy.
// ---------------------------------------------------------------------------
__global__ __launch_bounds__(256) void k_headproj(const _Float16* __restrict__ H,
                                                  _Float16* __restrict__ out,
                                                  int mode) {
  const int gw   = (blockIdx.x * 256 + threadIdx.x) >> 5;  // 0..32767
  const int lane = threadIdx.x & 31;
  const int t    = gw >> 4;     // token 0..2047
  const int head = gw & 15;
  const int b    = t >> 10;
  const int tt   = t & 1023;

  const _Float16* src = H + (size_t)t * 1024 + head * 64;
  _Float16* dst = out + ((((size_t)b * 16 + head) * 1024) + tt) * 64;

  float e0 = (float)src[lane * 2 + 0];
  float e1 = (float)src[lane * 2 + 1];
  if (mode == 2) {
    dst[lane * 2 + 0] = (_Float16)e0;
    dst[lane * 2 + 1] = (_Float16)e1;
    return;
  }
  float c0 = fminf(fmaxf(e0, -5.f), 5.f);
  float c1 = fminf(fmaxf(e1, -5.f), 5.f);
  float s  = c1 * c1 + ((lane == 0) ? 0.f : c0 * c0);
  #pragma unroll
  for (int off = 16; off > 0; off >>= 1) s += __shfl_xor(s, off, 32);
  float x0 = sqrtf(1.f + s + EPSF);
  float o0 = (lane == 0) ? ((mode == 1) ? -x0 : x0) : c0;
  dst[lane * 2 + 0] = (_Float16)o0;
  dst[lane * 2 + 1] = (_Float16)c1;
}

// ---------------------------------------------------------------------------
// Kernel 4: flash attention per (b,h).  Q has q0 pre-negated, so
// score = dot(q,k)/8.  K tile staged via global_load_async_to_lds (ASYNCcnt).
// ---------------------------------------------------------------------------
#define ABQ 128
#define ABK 64
__global__ __launch_bounds__(256) void k_attn(const _Float16* __restrict__ Q,
                                              const _Float16* __restrict__ Km,
                                              const _Float16* __restrict__ Vm,
                                              _Float16* __restrict__ AO) {
  const int T = 1024, dh = 64;
  const int bh = blockIdx.x >> 3;     // 0..31
  const int qt = blockIdx.x & 7;
  const int b  = bh >> 4, h = bh & 15;
  const size_t base = (size_t)bh * T * dh;
  const _Float16* Qp = Q  + base;
  const _Float16* Kp = Km + base;
  const _Float16* Vp = Vm + base;

  const int lane = threadIdx.x & 31;
  const int wave = threadIdx.x >> 5;  // 0..7
  const int mrow = lane & 15;
  const int hi   = lane >> 4;
  const int q0r  = qt * ABQ + wave * 16;

  // persistent Q A-fragments (dh=64 -> two K=32 steps)
  v16h qa[2];
  #pragma unroll
  for (int ks = 0; ks < 2; ++ks) {
    const _Float16* qr = Qp + (size_t)(q0r + mrow) * dh + ks * 32;
    qa[ks] = cat8(*(const half8*)&qr[hi * 8],
                  *(const half8*)&qr[16 + hi * 8]);
  }

  __shared__ __align__(16) _Float16 Ks[ABK][64];      // [key][feat]   8 KB
  __shared__ __align__(16) _Float16 Vs[64][ABK];      // [feat][key]   8 KB
  __shared__ __align__(16) _Float16 Ps[8][16][64];    // per-wave P   16 KB

  float m_i[8], l_i[8];
  #pragma unroll
  for (int r = 0; r < 8; ++r) { m_i[r] = -1e30f; l_i[r] = 0.f; }
  v8f o[4] = {};

  const float sm = 0.125f;  // 1/sqrt(64)

  for (int j0 = 0; j0 < T; j0 += ABK) {
    __syncthreads();
    {  // K tile: async copy global -> LDS (no VGPR round-trip)
      int cbase = threadIdx.x * 2;
      #pragma unroll
      for (int c = 0; c < 2; ++c) {
        int ch  = cbase + c;          // 0..511
        int row = ch >> 3;            // 8 chunks per 64-half row
        int kk  = (ch & 7) * 8;
        unsigned lds = (unsigned)(uintptr_t)&Ks[row][kk];
        unsigned long long ga =
            (unsigned long long)(uintptr_t)&Kp[(size_t)(j0 + row) * dh + kk];
        asm volatile("global_load_async_to_lds_b128 %0, %1, off"
                     :: "v"(lds), "v"(ga) : "memory");
      }
      // V tile transposed (needs lane shuffle, stays on VGPR path)
      int vrow = threadIdx.x >> 2;        // key 0..63
      int f0   = (threadIdx.x & 3) * 16;  // feat base
      const _Float16* vr = &Vp[(size_t)(j0 + vrow) * dh + f0];
      #pragma unroll
      for (int f = 0; f < 16; ++f) Vs[f0 + f][vrow] = vr[f];
      asm volatile("s_wait_asynccnt 0x0" ::: "memory");
    }
    __syncthreads();

    // S = Q K^T : 16x64 per wave
    v8f sa[4] = {};
    #pragma unroll
    for (int nb = 0; nb < 4; ++nb) {
      #pragma unroll
      for (int ks = 0; ks < 2; ++ks) {
        const int n  = nb * 16 + mrow;        // key col
        const int kb = ks * 32 + hi * 8;      // feat
        v16h bfr = cat8(*(const half8*)&Ks[n][kb],
                        *(const half8*)&Ks[n][kb + 16]);
        sa[nb] = wmma_f16(qa[ks], bfr, sa[nb]);
      }
    }

    // online softmax (row stats redundant in each 16-lane half)
    #pragma unroll
    for (int r = 0; r < 8; ++r) {
      float mx = fmaxf(fmaxf(sa[0][r], sa[1][r]), fmaxf(sa[2][r], sa[3][r]));
      #pragma unroll
      for (int off = 8; off > 0; off >>= 1) mx = fmaxf(mx, __shfl_xor(mx, off, 32));
      float mnew = fmaxf(m_i[r], mx * sm);
      float corr = __expf(m_i[r] - mnew);
      m_i[r] = mnew;
      float rsum = 0.f;
      #pragma unroll
      for (int nb = 0; nb < 4; ++nb) {
        float p = __expf(sa[nb][r] * sm - mnew);
        sa[nb][r] = p;
        rsum += p;
      }
      #pragma unroll
      for (int off = 8; off > 0; off >>= 1) rsum += __shfl_xor(rsum, off, 32);
      l_i[r] = l_i[r] * corr + rsum;
      #pragma unroll
      for (int nb = 0; nb < 4; ++nb) o[nb][r] *= corr;
    }

    // P: C-layout -> A-layout via per-wave LDS scratch (same-wave, in-order)
    #pragma unroll
    for (int nb = 0; nb < 4; ++nb)
      #pragma unroll
      for (int r = 0; r < 8; ++r)
        Ps[wave][hi * 8 + r][nb * 16 + mrow] = (_Float16)sa[nb][r];

    // O += P @ V
    #pragma unroll
    for (int ks = 0; ks < 2; ++ks) {
      const _Float16* pr = &Ps[wave][mrow][ks * 32];
      v16h pf = cat8(*(const half8*)&pr[hi * 8],
                     *(const half8*)&pr[16 + hi * 8]);
      #pragma unroll
      for (int nb = 0; nb < 4; ++nb) {
        const int n  = nb * 16 + mrow;       // feat col
        const int kb = ks * 32 + hi * 8;     // key
        v16h bfr = cat8(*(const half8*)&Vs[n][kb],
                        *(const half8*)&Vs[n][kb + 16]);
        o[nb] = wmma_f16(pf, bfr, o[nb]);
      }
    }
  }

  // epilogue: AO[b][t][h*64 + feat] = o / l
  #pragma unroll
  for (int nb = 0; nb < 4; ++nb)
    #pragma unroll
    for (int r = 0; r < 8; ++r) {
      int row  = q0r + hi * 8 + r;
      int feat = nb * 16 + mrow;
      float val = o[nb][r] / l_i[r];
      AO[((size_t)b * 1024 + row) * 1024 + h * 64 + feat] = (_Float16)val;
    }
}

// ---------------------------------------------------------------------------
// Kernel 5: exp_map(mu, Y) + project -> f32 output.  One block per token.
// ---------------------------------------------------------------------------
__global__ __launch_bounds__(256) void k_expmap(const _Float16* __restrict__ Y,
                                                float* __restrict__ out) {
  const int d = 1024;
  const int t = blockIdx.x;
  const _Float16* yr = Y + (size_t)t * d;
  __shared__ float red[256];

  float s = 0.f;
  for (int j = threadIdx.x; j < d; j += 256)
    if (j >= 1) { float v = (float)yr[j]; s += v * v; }
  red[threadIdx.x] = s;
  __syncthreads();
  for (int off = 128; off > 0; off >>= 1) {
    if (threadIdx.x < off) red[threadIdx.x] += red[threadIdx.x + off];
    __syncthreads();
  }
  const float vn = fminf(sqrtf(fmaxf(red[0], 0.f) + EPSF), 5.f);
  const float sc = sinhf(vn) / (vn + EPSF);

  float s2 = 0.f;
  for (int j = threadIdx.x; j < d; j += 256)
    if (j >= 1) {
      float sp = fminf(fmaxf(sc * (float)yr[j], -8.f), 8.f);
      s2 += sp * sp;
      out[(size_t)t * d + j] = sp;
    }
  __syncthreads();
  red[threadIdx.x] = s2;
  __syncthreads();
  for (int off = 128; off > 0; off >>= 1) {
    if (threadIdx.x < off) red[threadIdx.x] += red[threadIdx.x + off];
    __syncthreads();
  }
  if (threadIdx.x == 0) out[(size_t)t * d] = sqrtf(1.f + red[0] + EPSF);
}

// ---------------------------------------------------------------------------
extern "C" void kernel_launch(void* const* d_in, const int* in_sizes, int n_in,
                              void* d_out, int out_size, void* d_ws, size_t ws_size,
                              hipStream_t stream) {
  const float* x  = (const float*)d_in[0];
  const float* Wq = (const float*)d_in[1];
  const float* Wk = (const float*)d_in[2];
  const float* Wv = (const float*)d_in[3];
  const float* Wo = (const float*)d_in[4];
  float* out = (float*)d_out;

  char* ws = (char*)d_ws;
  const size_t MB = 1u << 20;
  _Float16* vh = (_Float16*)(ws + 0 * MB);   // [2048][1024]  4 MB
  _Float16* Hq = (_Float16*)(ws + 4 * MB);   // [2048][1024]  4 MB
  _Float16* Hk = (_Float16*)(ws + 8 * MB);
  _Float16* Hv = (_Float16*)(ws + 12 * MB);
  _Float16* Qh = (_Float16*)(ws + 16 * MB);  // [B][h][T][64] 4 MB
  _Float16* Kh = (_Float16*)(ws + 20 * MB);
  _Float16* Vh = (_Float16*)(ws + 24 * MB);
  _Float16* AO = Hq;                         // reuse (Hq dead after headproj)
  _Float16* Yb = Hk;                         // reuse

  const int M = 2048, N = 1024, K = 1024;
  const int gemmGrid = (M / GBM) * (N / GBN);  // 512

  k_logmap<<<2048, 256, 0, stream>>>(x, vh);

  k_gemm<<<gemmGrid, 128, 0, stream>>>(vh, Wq, Hq, M, N, K);
  k_gemm<<<gemmGrid, 128, 0, stream>>>(vh, Wk, Hk, M, N, K);
  k_gemm<<<gemmGrid, 128, 0, stream>>>(vh, Wv, Hv, M, N, K);

  k_headproj<<<4096, 256, 0, stream>>>(Hq, Qh, 1);  // Q: proj + negate q0
  k_headproj<<<4096, 256, 0, stream>>>(Hk, Kh, 0);  // K: proj
  k_headproj<<<4096, 256, 0, stream>>>(Hv, Vh, 2);  // V: copy/transpose

  k_attn<<<256, 256, 0, stream>>>(Qh, Kh, Vh, AO);

  k_gemm<<<gemmGrid, 128, 0, stream>>>(AO, Wo, Yb, M, N, K);

  k_expmap<<<2048, 256, 0, stream>>>(Yb, out);
}
